// DecoderRNN_47820165873666
// MI455X (gfx1250) — matrix operand order repaired and checked by reading, not compiled, not run on hardware
//
#include <hip/hip_runtime.h>
#include <hip/hip_bf16.h>
#include <math.h>

typedef __attribute__((ext_vector_type(16))) __bf16 v16bf;
typedef __attribute__((ext_vector_type(8)))  float  v8f;

#define Bsz 64
#define Pn  196
#define ENC 2048
#define Ee  512
#define Hh  512
#define Vv  10000
#define Tt  20
#define LDSLD 520   // 512 + 8 bf16 pad: row stride 260 dwords -> conflict-free b128 reads

// ---------------------------------------------------------------------------
// WMMA fragment layouts (wave32, 16x16x32 bf16):
//  A 16x32 : lane&15 = row M; lanes 0-15 K0-15, lanes 16-31 K16-31 (k order chosen here, not AMD's table order)
//  B 32x16 : lane&15 = col N; lanes 0-15 K0-15, lanes 16-31 K16-31 (same chosen k order as A)
//  C/D     : VGPR j -> M = j + 8*(lane>>4), N = lane&15
// ---------------------------------------------------------------------------

// f32 convert-on-load tile (one-time GEMMs). Arow/Wrow are per-lane row ptrs.
__device__ __forceinline__ v8f wmma_acc_tile_f32(const float* __restrict__ Arow,
                                                 const float* __restrict__ Wrow,
                                                 int K, v8f acc, int lane) {
  const int hi = lane >> 4;
  for (int k0 = 0; k0 < K; k0 += 32) {
    v16bf a, b;
#pragma unroll
    for (int j = 0; j < 16; ++j) {
      a[j] = (__bf16)Arow[k0 + j + hi * 16];
      b[j] = (__bf16)Wrow[k0 + j + hi * 16];
    }
    acc = __builtin_amdgcn_wmma_f32_16x16x32_bf16(false, a, false, b,
                                                  (short)0, acc, false, false);
  }
  return acc;
}

// bf16 path, one k-chunk, 4 M-tiles (rows 0..63) sharing one B fragment.
// A: LDS-staged 64-row activation tile (ld = lda). Wrow: per-lane bf16 weight row.
__device__ __forceinline__ void wmma_step4(const __bf16* A, int lda,
                                           const __bf16* __restrict__ Wrow,
                                           int k0, int lane, v8f* acc) {
  const int l = lane & 15, hi = lane >> 4;
  v16bf b;
#pragma unroll
  for (int j = 0; j < 16; ++j) b[j] = Wrow[k0 + j + hi * 16];
#pragma unroll
  for (int mt = 0; mt < 4; ++mt) {
    const __bf16* Ar = A + (mt * 16 + l) * lda;
    v16bf a;
#pragma unroll
    for (int j = 0; j < 16; ++j) a[j] = Ar[k0 + j + hi * 16];
    acc[mt] = __builtin_amdgcn_wmma_f32_16x16x32_bf16(false, a, false, b,
                                                      (short)0, acc[mt], false, false);
  }
}

// Cooperative 64x512-bf16 stage into LDS (row stride LDSLD), 16B vector moves.
__device__ __forceinline__ void stage_rows(__bf16* dst, const __bf16* __restrict__ src,
                                           int src_ld) {
  for (int i = threadIdx.x; i < 64 * 64; i += 256) {
    int r = i >> 6, cc = (i & 63) * 8;
    *(uint4*)(dst + r * LDSLD + cc) = *(const uint4*)(src + (size_t)r * src_ld + cc);
  }
}

__device__ __forceinline__ float sigf(float x) { return 1.0f / (1.0f + expf(-x)); }

// ---------------------------------------------------------------------------
// Pre-pass kernels (timestep-invariant work)
// ---------------------------------------------------------------------------
__global__ __launch_bounds__(256) void mean_kernel(const float* __restrict__ img,
                                                   float* __restrict__ avg) {
  int e = blockIdx.x * 256 + threadIdx.x;
  int b = blockIdx.y;
  const float* base = img + (size_t)b * Pn * ENC + e;
  float s = 0.f;
  for (int p = 0; p < Pn; ++p) s += base[(size_t)p * ENC];
  avg[(size_t)b * ENC + e] = s * (1.0f / (float)Pn);
}

__global__ __launch_bounds__(256) void weff_kernel(const float* __restrict__ W_enc_att,
                                                   const float* __restrict__ W_full,
                                                   float* __restrict__ w_eff) {
  int e = blockIdx.x * 256 + threadIdx.x;
  float s = 0.f;
  for (int a = 0; a < 512; ++a) s += W_full[a] * W_enc_att[(size_t)a * ENC + e];
  w_eff[e] = s;
}

// alpha[b,p] = softmax_p( img[b,p,:] . w_eff )  (shift-invariance: exact vs ref)
__global__ __launch_bounds__(256) void alpha_kernel(const float* __restrict__ img,
                                                    const float* __restrict__ w_eff,
                                                    float* __restrict__ alpha_ws,
                                                    float* __restrict__ out_alphas) {
  __shared__ float att_s[Pn];
  int b = blockIdx.x;
  int lane = threadIdx.x & 31, wv = threadIdx.x >> 5;
  const float* base = img + (size_t)b * Pn * ENC;
  for (int p = wv; p < Pn; p += 8) {
    const float* row = base + (size_t)p * ENC;
    float s = 0.f;
    for (int k = lane; k < ENC; k += 32) s += row[k] * w_eff[k];
    for (int off = 16; off; off >>= 1) s += __shfl_xor(s, off, 32);
    if (lane == 0) att_s[p] = s;
  }
  __syncthreads();
  if (wv == 0) {
    float m = -1e30f;
    for (int p = lane; p < Pn; p += 32) m = fmaxf(m, att_s[p]);
    for (int off = 16; off; off >>= 1) m = fmaxf(m, __shfl_xor(m, off, 32));
    float s = 0.f;
    for (int p = lane; p < Pn; p += 32) { float e = expf(att_s[p] - m); att_s[p] = e; s += e; }
    for (int off = 16; off; off >>= 1) s += __shfl_xor(s, off, 32);
    float inv = 1.0f / s;
    for (int p = lane; p < Pn; p += 32) att_s[p] *= inv;
  }
  __syncthreads();
  for (int p = threadIdx.x; p < Pn; p += 256) {
    float al = att_s[p];
    alpha_ws[(size_t)b * Pn + p] = al;
    for (int t = 0; t < Tt; ++t)
      out_alphas[((size_t)b * Tt + t) * Pn + p] = al;
  }
}

__global__ __launch_bounds__(256) void context_kernel(const float* __restrict__ img,
                                                      const float* __restrict__ alpha,
                                                      float* __restrict__ context) {
  int e = blockIdx.x * 256 + threadIdx.x;
  int b = blockIdx.y;
  const float* base = img + (size_t)b * Pn * ENC + e;
  const float* al = alpha + (size_t)b * Pn;
  float s = 0.f;
  for (int p = 0; p < Pn; ++p) s += base[(size_t)p * ENC] * al[p];
  context[(size_t)b * ENC + e] = s;
}

// Strided f32 -> bf16 weight repack. cols is a power of two (passed as log2).
__global__ __launch_bounds__(256) void cvt_bf16_kernel(const float* __restrict__ src,
                                                       __bf16* __restrict__ dst,
                                                       int log2cols, int src_ld,
                                                       int src_off, size_t total) {
  size_t idx = (size_t)blockIdx.x * 256 + threadIdx.x;
  size_t stride = (size_t)gridDim.x * 256;
  int cmask = (1 << log2cols) - 1;
  for (; idx < total; idx += stride) {
    size_t r = idx >> log2cols;
    int cc = (int)(idx & cmask);
    dst[idx] = (__bf16)src[r * src_ld + src_off + cc];
  }
}

// h0/c0 = avg @ W_init.T + b (M=64, N=512, K=2048); blockIdx.y: 0 -> h(bf16), 1 -> c(f32)
__global__ __launch_bounds__(256) void init_hc_kernel(const float* __restrict__ avg,
                                                      const float* __restrict__ Wh,
                                                      const float* __restrict__ bh,
                                                      const float* __restrict__ Wc,
                                                      const float* __restrict__ bc,
                                                      __bf16* __restrict__ h_bf,
                                                      float* __restrict__ c) {
  int lane = threadIdx.x & 31, w = threadIdx.x >> 5;
  int l = lane & 15, hi = lane >> 4;
  int m0 = (w & 3) * 16;
  int n0 = blockIdx.x * 32 + (w >> 2) * 16;
  const float* W    = blockIdx.y ? Wc : Wh;
  const float* bias = blockIdx.y ? bc : bh;
  v8f acc = {};
  acc = wmma_acc_tile_f32(avg + (size_t)(m0 + l) * ENC, W + (size_t)(n0 + l) * ENC,
                          ENC, acc, lane);
  int n = n0 + l;
  float bv = bias[n];
  if (blockIdx.y == 0) {
#pragma unroll
    for (int j = 0; j < 8; ++j)
      h_bf[(size_t)(m0 + j + 8 * hi) * Hh + n] = (__bf16)(acc[j] + bv);
  } else {
#pragma unroll
    for (int j = 0; j < 8; ++j)
      c[(size_t)(m0 + j + 8 * hi) * Hh + n] = acc[j] + bv;
  }
}

// premix[t*64+b, n] = emb(captions[b,t]) @ W_ih[:, :E].T + b_ih + b_hh
// M = T*B = 1280, N = 2048, K = 512
__global__ __launch_bounds__(256) void premix_kernel(const float* __restrict__ emb_table,
                                                     const int* __restrict__ captions,
                                                     const float* __restrict__ W_ih,
                                                     const float* __restrict__ b_ih,
                                                     const float* __restrict__ b_hh,
                                                     float* __restrict__ premix) {
  int lane = threadIdx.x & 31, w = threadIdx.x >> 5;
  int l = lane & 15, hi = lane >> 4;
  int m0 = blockIdx.y * 32 + (w & 1) * 16;
  int n0 = blockIdx.x * 64 + (w >> 1) * 16;
  int row = m0 + l;
  int t = row >> 6, b = row & 63;
  int tok = captions[b * Tt + t];
  v8f acc = {};
  acc = wmma_acc_tile_f32(emb_table + (size_t)tok * Ee,
                          W_ih + (size_t)(n0 + l) * (Ee + ENC), Ee, acc, lane);
  int n = n0 + l;
  float bv = b_ih[n] + b_hh[n];
#pragma unroll
  for (int j = 0; j < 8; ++j)
    premix[(size_t)(m0 + j + 8 * hi) * 2048 + n] = acc[j] + bv;
}

// ---------------------------------------------------------------------------
// Per-timestep kernels: one wave = one 16-col N-tile x all 64 rows (4 acc).
// Activations staged once per block in LDS; weights streamed once per wave.
// ---------------------------------------------------------------------------
// gx = sigmoid(h @ W_beta.T + b_beta) * context  (M=64, N=2048, K=512)
__global__ __launch_bounds__(256) void gate_gx_kernel(const __bf16* __restrict__ h_bf,
                                                      const __bf16* __restrict__ Wb_bf,
                                                      const float* __restrict__ b_beta,
                                                      const float* __restrict__ context,
                                                      __bf16* __restrict__ gx_bf) {
  __shared__ __bf16 sh[64 * LDSLD];
  int lane = threadIdx.x & 31, w = threadIdx.x >> 5;
  int l = lane & 15, hi = lane >> 4;
  int n0 = (blockIdx.x * 8 + w) * 16;            // grid 16 -> N=2048
  stage_rows(sh, h_bf, Hh);
  __syncthreads();
  const __bf16* Wrow = Wb_bf + (size_t)(n0 + l) * Hh;
  v8f acc[4] = {};
  for (int k0 = 0; k0 < Hh; k0 += 32) wmma_step4(sh, LDSLD, Wrow, k0, lane, acc);
  int n = n0 + l;
  float bv = b_beta[n];
#pragma unroll
  for (int mt = 0; mt < 4; ++mt)
#pragma unroll
    for (int j = 0; j < 8; ++j) {
      int r = mt * 16 + j + 8 * hi;
      float g = sigf(acc[mt][j] + bv);
      gx_bf[(size_t)r * ENC + n] = (__bf16)(g * context[(size_t)r * ENC + n]);
    }
}

// gates = premix_t + gx @ W_ih[:, E:].T + h @ W_hh.T  (M=64, N=2048)
__global__ __launch_bounds__(256) void gates_kernel(const __bf16* __restrict__ gx_bf,
                                                    const __bf16* __restrict__ h_bf,
                                                    const __bf16* __restrict__ Wihe_bf,
                                                    const __bf16* __restrict__ Whh_bf,
                                                    const float* __restrict__ premix_t,
                                                    float* __restrict__ gates) {
  __shared__ __bf16 sh[64 * LDSLD];
  int lane = threadIdx.x & 31, w = threadIdx.x >> 5;
  int l = lane & 15, hi = lane >> 4;
  int n0 = (blockIdx.x * 8 + w) * 16;            // grid 16 -> N=2048
  v8f acc[4] = {};
  const __bf16* Wrow1 = Wihe_bf + (size_t)(n0 + l) * ENC;
  for (int kb = 0; kb < 4; ++kb) {               // K=2048 staged in 512-chunks
    if (kb) __syncthreads();
    stage_rows(sh, gx_bf + kb * 512, ENC);
    __syncthreads();
    const __bf16* W1 = Wrow1 + kb * 512;
    for (int k0 = 0; k0 < 512; k0 += 32) wmma_step4(sh, LDSLD, W1, k0, lane, acc);
  }
  __syncthreads();
  stage_rows(sh, h_bf, Hh);
  __syncthreads();
  const __bf16* Wrow2 = Whh_bf + (size_t)(n0 + l) * Hh;
  for (int k0 = 0; k0 < Hh; k0 += 32) wmma_step4(sh, LDSLD, Wrow2, k0, lane, acc);
  int n = n0 + l;
#pragma unroll
  for (int mt = 0; mt < 4; ++mt)
#pragma unroll
    for (int j = 0; j < 8; ++j) {
      int r = mt * 16 + j + 8 * hi;
      gates[(size_t)r * 2048 + n] = acc[mt][j] + premix_t[(size_t)r * 2048 + n];
    }
}

// LSTM pointwise: h (bf16) and c (f32), 64x512
__global__ __launch_bounds__(256) void lstm_kernel(const float* __restrict__ gates,
                                                   __bf16* __restrict__ h_bf,
                                                   float* __restrict__ c) {
  int idx = blockIdx.x * 256 + threadIdx.x;
  int b = idx >> 9, j = idx & 511;
  const float* g = gates + (size_t)b * 2048;
  float ig = sigf(g[j]);
  float fg = sigf(g[512 + j]);
  float gg = tanhf(g[1024 + j]);
  float og = sigf(g[1536 + j]);
  float cn = fg * c[idx] + ig * gg;
  c[idx] = cn;
  h_bf[idx] = (__bf16)(og * tanhf(cn));
}

// preds[b,t,:] = h @ W_out.T + b_out  (M=64, N=10000, K=512)
__global__ __launch_bounds__(256) void pred_kernel(const __bf16* __restrict__ h_bf,
                                                   const __bf16* __restrict__ Wout_bf,
                                                   const float* __restrict__ b_out,
                                                   float* __restrict__ preds, int t) {
  __shared__ __bf16 sh[64 * LDSLD];
  int lane = threadIdx.x & 31, w = threadIdx.x >> 5;
  int l = lane & 15, hi = lane >> 4;
  int n0 = (blockIdx.x * 8 + w) * 16;            // 625 tiles, grid 79
  stage_rows(sh, h_bf, Hh);
  __syncthreads();
  if (n0 >= Vv) return;                          // wave-uniform guard (after barrier)
  const __bf16* Wrow = Wout_bf + (size_t)(n0 + l) * Hh;
  v8f acc[4] = {};
  for (int k0 = 0; k0 < Hh; k0 += 32) wmma_step4(sh, LDSLD, Wrow, k0, lane, acc);
  int n = n0 + l;
  float bv = b_out[n];
#pragma unroll
  for (int mt = 0; mt < 4; ++mt)
#pragma unroll
    for (int j = 0; j < 8; ++j) {
      int r = mt * 16 + j + 8 * hi;
      preds[((size_t)r * Tt + t) * Vv + n] = acc[mt][j] + bv;
    }
}

// ---------------------------------------------------------------------------
extern "C" void kernel_launch(void* const* d_in, const int* in_sizes, int n_in,
                              void* d_out, int out_size, void* d_ws, size_t ws_size,
                              hipStream_t stream) {
  const float* img       = (const float*)d_in[0];
  const int*   captions  = (const int*)d_in[1];
  const float* embedding = (const float*)d_in[2];
  const float* W_enc_att = (const float*)d_in[3];
  // d_in[4] b_enc_att, [5] W_dec_att, [6] b_dec_att, [8] b_full cancel exactly
  // under softmax shift-invariance.
  const float* W_full    = (const float*)d_in[7];
  const float* W_beta    = (const float*)d_in[9];
  const float* b_beta    = (const float*)d_in[10];
  const float* W_ih      = (const float*)d_in[11];
  const float* b_ih      = (const float*)d_in[12];
  const float* W_hh      = (const float*)d_in[13];
  const float* b_hh      = (const float*)d_in[14];
  const float* W_init_h  = (const float*)d_in[15];
  const float* b_init_h  = (const float*)d_in[16];
  const float* W_init_c  = (const float*)d_in[17];
  const float* b_init_c  = (const float*)d_in[18];
  const float* W_out     = (const float*)d_in[19];
  const float* b_out     = (const float*)d_in[20];

  float* preds      = (float*)d_out;                 // [64,20,10000]
  float* out_alphas = preds + (size_t)Bsz * Tt * Vv; // [64,20,196]

  // ---- workspace: f32 region, then bf16 region (16B-aligned boundaries) ----
  float* avg     = (float*)d_ws;                      // 64*2048
  float* w_eff   = avg     + (size_t)Bsz * ENC;       // 2048
  float* alpha   = w_eff   + ENC;                     // 64*196
  float* context = alpha   + (size_t)Bsz * Pn;        // 64*2048
  float* c       = context + (size_t)Bsz * ENC;       // 64*512
  float* premix  = c       + (size_t)Bsz * Hh;        // 20*64*2048
  float* gates   = premix  + (size_t)Tt * Bsz * 2048; // 64*2048
  __bf16* h_bf    = (__bf16*)(gates + (size_t)Bsz * 2048);
  __bf16* gx_bf   = h_bf    + (size_t)Bsz * Hh;        // 64*2048
  __bf16* Wb_bf   = gx_bf   + (size_t)Bsz * ENC;       // 2048*512
  __bf16* Wihe_bf = Wb_bf   + (size_t)ENC * Hh;        // 2048*2048
  __bf16* Whh_bf  = Wihe_bf + (size_t)ENC * ENC;       // 2048*512
  __bf16* Wout_bf = Whh_bf  + (size_t)ENC * Hh;        // 10000*512

  // ---- timestep-invariant pre-pass ----
  mean_kernel   <<<dim3(ENC / 256, Bsz), 256, 0, stream>>>(img, avg);
  weff_kernel   <<<ENC / 256, 256, 0, stream>>>(W_enc_att, W_full, w_eff);
  alpha_kernel  <<<Bsz, 256, 0, stream>>>(img, w_eff, alpha, out_alphas);
  context_kernel<<<dim3(ENC / 256, Bsz), 256, 0, stream>>>(img, alpha, context);
  init_hc_kernel<<<dim3(Hh / 32, 2), 256, 0, stream>>>(avg, W_init_h, b_init_h,
                                                       W_init_c, b_init_c, h_bf, c);
  premix_kernel <<<dim3(2048 / 64, (Tt * Bsz) / 32), 256, 0, stream>>>(
      embedding, captions, W_ih, b_ih, b_hh, premix);

  // ---- one-time weight repack to bf16 (reused 20x in the recurrence) ----
  cvt_bf16_kernel<<<1024, 256, 0, stream>>>(W_beta, Wb_bf,   9,  Hh,       0,
                                            (size_t)ENC * Hh);
  cvt_bf16_kernel<<<2048, 256, 0, stream>>>(W_ih,   Wihe_bf, 11, Ee + ENC, Ee,
                                            (size_t)ENC * ENC);
  cvt_bf16_kernel<<<1024, 256, 0, stream>>>(W_hh,   Whh_bf,  9,  Hh,       0,
                                            (size_t)ENC * Hh);
  cvt_bf16_kernel<<<2048, 256, 0, stream>>>(W_out,  Wout_bf, 9,  Hh,       0,
                                            (size_t)Vv * Hh);

  // ---- recurrent steps ----
  for (int t = 0; t < Tt; ++t) {
    gate_gx_kernel<<<ENC / 128, 256, 0, stream>>>(h_bf, Wb_bf, b_beta, context, gx_bf);
    gates_kernel  <<<2048 / 128, 256, 0, stream>>>(gx_bf, h_bf, Wihe_bf, Whh_bf,
                                                   premix + (size_t)t * Bsz * 2048, gates);
    lstm_kernel   <<<(Bsz * Hh) / 256, 256, 0, stream>>>(gates, h_bf, c);
    pred_kernel   <<<(Vv / 16 + 7) / 8, 256, 0, stream>>>(h_bf, Wout_bf, b_out, preds, t);
  }
}
